// DependencyTreeModel_75857712382248
// MI455X (gfx1250) — matrix operand order repaired and checked by reading, not compiled, not run on hardware
//
#include <hip/hip_runtime.h>
#include <hip/hip_bf16.h>
#include <math.h>

typedef __attribute__((ext_vector_type(16))) _Float16 v16h;
typedef __attribute__((ext_vector_type(8)))  float    v8f;
typedef __attribute__((ext_vector_type(2)))  float    v2f;

#define B_ 8
#define N_ 1024
#define H_ 256

// ---------------- WMMA fragment loaders (CDNA5 wave32, contiguous k = 16*half + e in A and B) ----------------
__device__ __forceinline__ int lane_id() { return (int)(threadIdx.x & 31); }

// A matrix 16x32 f16 (MxK), row-major src, leading dim ld.
__device__ __forceinline__ v16h load_a_frag(const _Float16* src, int ld, int row0, int kbase) {
  const int lane = lane_id();
  const int m    = lane & 15;
  const int half = lane >> 4;
  const _Float16* p = src + (size_t)(row0 + m) * ld;
  const int kA = kbase + half * 16;
  v16h a;
#pragma unroll
  for (int e = 0; e < 16; ++e) a[e] = p[kA + e];
  return a;
}

// B matrix 32x16 f16 (KxN), element (K,n) = src[(n0+n)*ld + kbase+K]  -> contiguous 32B per lane
__device__ __forceinline__ v16h load_b_frag_trans(const _Float16* src, int ld, int n0, int kbase) {
  const int lane = lane_id();
  const int n  = lane & 15;
  const int kb = kbase + ((lane >> 4) << 4);
  const _Float16* p = src + (size_t)(n0 + n) * ld + kb;
  v16h b;
#pragma unroll
  for (int e = 0; e < 16; ++e) b[e] = p[e];
  return b;
}

// ---------------- Kernels ----------------

__global__ void f32_to_f16_kernel(const float* __restrict__ src, _Float16* __restrict__ dst, int count) {
  int idx = blockIdx.x * blockDim.x + threadIdx.x;
  if (idx < count) dst[idx] = (_Float16)src[idx];
}

// wb16T[k][g][h] = (f16) W_bilin[k][h][g]  -- pre-transpose so GEMM B-fragments are K-contiguous
__global__ void wbilin_transpose_f16_kernel(const float* __restrict__ src, _Float16* __restrict__ dst) {
  int idx = blockIdx.x * blockDim.x + threadIdx.x;   // exactly 2*H_*H_ threads
  int k   = idx >> 16;
  int rem = idx & 65535;
  int g   = rem >> 8;
  int h   = rem & 255;
  dst[((size_t)k * H_ + g) * H_ + h] = (_Float16)src[((size_t)k * H_ + h) * H_ + g];
}

// tmp[b,kk,i,g] = sum_h h[b,i,h] * W_bilin[kk,h,g]   (one 16x16 tile per wave)
__global__ void tmp_gemm_kernel(const _Float16* __restrict__ h16,
                                const _Float16* __restrict__ wb16T,
                                _Float16* __restrict__ tmp16) {
  int wid  = blockIdx.x * (blockDim.x >> 5) + (threadIdx.x >> 5);
  int b    = wid >> 11;
  int rem  = wid & 2047;
  int kk   = rem >> 10;
  int rem2 = rem & 1023;
  int i0   = (rem2 >> 4) << 4;
  int g0   = (rem2 & 15) << 4;
  const _Float16* A = h16   + (size_t)b  * N_ * H_;
  const _Float16* W = wb16T + (size_t)kk * H_ * H_;
  v8f c = {};
#pragma unroll
  for (int kb = 0; kb < H_; kb += 32) {
    v16h a = load_a_frag(A, H_, i0, kb);
    v16h w = load_b_frag_trans(W, H_, g0, kb);
    c = __builtin_amdgcn_wmma_f32_16x16x32_f16(false, a, false, w, (short)0, c, false, false);
  }
  const int lane = lane_id();
  const int nn = lane & 15;
  const int mb = (lane >> 4) << 3;
  _Float16* dst = tmp16 + ((size_t)(b * 2 + kk) * N_ + i0) * H_ + g0;
#pragma unroll
  for (int r = 0; r < 8; ++r) dst[(size_t)(mb + r) * H_ + nn] = (_Float16)c[r];
}

// head/dep projections, one thread per (b,i)
__global__ void head_dep_kernel(const float* __restrict__ h_cat,
                                const float* __restrict__ W_head,
                                const float* __restrict__ W_dep,
                                float* __restrict__ head, float* __restrict__ dep) {
  int tid = blockIdx.x * blockDim.x + threadIdx.x;
  if (tid >= B_ * N_) return;
  const float* h = h_cat + (size_t)tid * H_;
#pragma unroll
  for (int kk = 0; kk < 2; ++kk) {
    float sh = 0.f, sd = 0.f;
    for (int g = 0; g < H_; ++g) {
      float hv = h[g];
      sh += hv * W_head[kk * H_ + g];
      sd += hv * W_dep[kk * H_ + g];
    }
    head[(size_t)tid * 2 + kk] = sh;
    dep[(size_t)tid * 2 + kk]  = sd;
  }
}

// root classifier: GELU(h @ W_r1^T + b_r1) @ W_r2^T + b_r2, fused via WMMA
__global__ void root_kernel(const _Float16* __restrict__ h16,
                            const _Float16* __restrict__ wr1,
                            const float* __restrict__ b_r1,
                            const float* __restrict__ W_r2,
                            const float* __restrict__ b_r2,
                            float* __restrict__ rootsc) {
  int wid = blockIdx.x * (blockDim.x >> 5) + (threadIdx.x >> 5);
  int b   = wid >> 6;
  int i0  = (wid & 63) << 4;
  const _Float16* Hb = h16 + (size_t)b * N_ * H_;
  const int lane = lane_id();
  const int nn = lane & 15;
  const int mb = (lane >> 4) << 3;
  float acc[8];
#pragma unroll
  for (int r = 0; r < 8; ++r) acc[r] = 0.f;
  for (int ct = 0; ct < 16; ++ct) {
    v8f c = {};
#pragma unroll
    for (int kb = 0; kb < H_; kb += 32) {
      v16h a = load_a_frag(Hb, H_, i0, kb);
      v16h w = load_b_frag_trans(wr1, H_, ct * 16, kb);
      c = __builtin_amdgcn_wmma_f32_16x16x32_f16(false, a, false, w, (short)0, c, false, false);
    }
    int col = ct * 16 + nn;
    float br = b_r1[col], w2 = W_r2[col];
#pragma unroll
    for (int r = 0; r < 8; ++r) {
      float x = c[r] + br;
      float g = 0.5f * x * (1.f + erff(x * 0.70710678118654752f));
      acc[r] += g * w2;
    }
  }
#pragma unroll
  for (int r = 0; r < 8; ++r) {
    float v = acc[r];
    for (int off = 8; off > 0; off >>= 1) v += __shfl_xor(v, off, 32);
    acc[r] = v;
  }
  if (nn == 0) {
    float br2 = b_r2[0];
#pragma unroll
    for (int r = 0; r < 8; ++r) rootsc[(size_t)b * N_ + i0 + mb + r] = acc[r] + br2;
  }
}

// compat: 32x32 output per wave (2 channels x 2x2 subtiles => 8 WMMA per K-step, b-frags shared)
__global__ void compat_kernel(const _Float16* __restrict__ tmp16,
                              const _Float16* __restrict__ h16,
                              const float* __restrict__ head,
                              const float* __restrict__ dep,
                              const float* __restrict__ b_bilin,
                              const float* __restrict__ left,
                              const float* __restrict__ right,
                              float* __restrict__ Aout,
                              float* __restrict__ partials) {
  int wid = blockIdx.x * (blockDim.x >> 5) + (threadIdx.x >> 5);
  int b   = wid >> 10;            // 32*32 = 1024 tiles of 32x32 per batch
  int rem = wid & 1023;
  int i0  = (rem >> 5) << 5;
  int j0  = (rem & 31) << 5;
  const _Float16* A0 = tmp16 + (size_t)(b * 2 + 0) * N_ * H_;
  const _Float16* A1 = tmp16 + (size_t)(b * 2 + 1) * N_ * H_;
  const _Float16* Hb = h16 + (size_t)b * N_ * H_;
  const int lane = lane_id();
  const int nn = lane & 15;
  const int mb = (lane >> 4) << 3;

  const float* Lb = left  + (size_t)b * N_ * N_;
  const float* Rb = right + (size_t)b * N_ * N_;
  // stream-prefetch adjacency rows (each 32-col row segment = one 128B line)
#pragma unroll
  for (int si = 0; si < 2; ++si) {
#pragma unroll
    for (int r = 0; r < 8; ++r) {
      size_t e = (size_t)(i0 + si * 16 + mb + r) * N_ + j0 + nn;
      __builtin_prefetch(&Lb[e], 0, 1);
      __builtin_prefetch(&Rb[e], 0, 1);
    }
  }

  v8f c00 = {}, c01 = {}, c10 = {}, c11 = {};
  v8f d00 = {}, d01 = {}, d10 = {}, d11 = {};
#pragma unroll
  for (int kb = 0; kb < H_; kb += 32) {
    v16h blo  = load_b_frag_trans(Hb, H_, j0,      kb);
    v16h bhi  = load_b_frag_trans(Hb, H_, j0 + 16, kb);
    v16h a0lo = load_a_frag(A0, H_, i0,      kb);
    v16h a0hi = load_a_frag(A0, H_, i0 + 16, kb);
    v16h a1lo = load_a_frag(A1, H_, i0,      kb);
    v16h a1hi = load_a_frag(A1, H_, i0 + 16, kb);
    c00 = __builtin_amdgcn_wmma_f32_16x16x32_f16(false, a0lo, false, blo, (short)0, c00, false, false);
    c01 = __builtin_amdgcn_wmma_f32_16x16x32_f16(false, a0lo, false, bhi, (short)0, c01, false, false);
    c10 = __builtin_amdgcn_wmma_f32_16x16x32_f16(false, a0hi, false, blo, (short)0, c10, false, false);
    c11 = __builtin_amdgcn_wmma_f32_16x16x32_f16(false, a0hi, false, bhi, (short)0, c11, false, false);
    d00 = __builtin_amdgcn_wmma_f32_16x16x32_f16(false, a1lo, false, blo, (short)0, d00, false, false);
    d01 = __builtin_amdgcn_wmma_f32_16x16x32_f16(false, a1lo, false, bhi, (short)0, d01, false, false);
    d10 = __builtin_amdgcn_wmma_f32_16x16x32_f16(false, a1hi, false, blo, (short)0, d10, false, false);
    d11 = __builtin_amdgcn_wmma_f32_16x16x32_f16(false, a1hi, false, bhi, (short)0, d11, false, false);
  }

  const float bb0 = b_bilin[0], bb1 = b_bilin[1];
  const float* headb = head + (size_t)b * N_ * 2;
  const float* depb  = dep  + (size_t)b * N_ * 2;
  float* Ab = Aout + (size_t)b * N_ * N_;
  float part = 0.f;

#define COMPAT_EPILOG(CC, DD, I0S, J0S)                                          \
  do {                                                                           \
    int gj = (J0S) + nn;                                                         \
    float d0 = depb[(size_t)gj * 2 + 0], d1 = depb[(size_t)gj * 2 + 1];          \
    _Pragma("unroll")                                                            \
    for (int r = 0; r < 8; ++r) {                                                \
      int gi = (I0S) + mb + r;                                                   \
      float v0 = CC[r] + headb[(size_t)gi * 2 + 0] + d0 + bb0;                   \
      float v1 = DD[r] + headb[(size_t)gi * 2 + 1] + d1 + bb1;                   \
      size_t e = (size_t)gi * N_ + gj;                                           \
      Ab[e] = expf(v0) + expf(v1);                                               \
      part += v0 * Lb[e] + v1 * Rb[e];                                           \
    }                                                                            \
  } while (0)

  COMPAT_EPILOG(c00, d00, i0,      j0);
  COMPAT_EPILOG(c01, d01, i0,      j0 + 16);
  COMPAT_EPILOG(c10, d10, i0 + 16, j0);
  COMPAT_EPILOG(c11, d11, i0 + 16, j0 + 16);
#undef COMPAT_EPILOG

  for (int off = 16; off > 0; off >>= 1) part += __shfl_xor(part, off, 32);
  if (lane == 0) partials[(size_t)b * 1024 + rem] = part;
}

// colsum[b][j] = sum_i A[b][i][j]
__global__ void colsum_kernel(const float* __restrict__ A, float* __restrict__ colsum) {
  int tid = blockIdx.x * blockDim.x + threadIdx.x;
  if (tid >= B_ * N_) return;
  int b = tid >> 10, j = tid & 1023;
  const float* Ab = A + (size_t)b * N_ * N_;
  float s = 0.f;
  for (int i = 0; i < N_; ++i) s += Ab[(size_t)i * N_ + j];
  colsum[tid] = s;
}

// in-place: lap = diag(colsum) - A ; row 0 = exp(root_scores)
__global__ void build_lap_kernel(float* __restrict__ lap,
                                 const float* __restrict__ colsum,
                                 const float* __restrict__ rootsc) {
  int idx = blockIdx.x * blockDim.x + threadIdx.x;
  int b   = idx >> 20;
  int rem = idx & ((1 << 20) - 1);
  int i   = rem >> 10;
  int j   = rem & 1023;
  float* Ab = lap + (size_t)b * N_ * N_;
  if (i == 0) {
    Ab[j] = expf(rootsc[(size_t)b * N_ + j]);
  } else {
    float v = -Ab[rem];
    if (i == j) v += colsum[(size_t)b * N_ + j];
    Ab[rem] = v;
  }
}

// Blocked LU (NB=16) with partial pivoting; one 1024-thread workgroup per batch.
// Panel: wave-shuffle pivot search (2 barriers), rank-1 updates restricted to panel.
// Block row: LDS-staged L11 triangular solve. Trailing: A22 -= L21*U12 via f32 WMMA 16x16x4.
__global__ void __launch_bounds__(1024) lu_kernel(float* __restrict__ lap,
                                                  float* __restrict__ logdetv,
                                                  float* __restrict__ signv) {
  const int n = N_;
  const int b = blockIdx.x;
  float* Ab = lap + (size_t)b * n * n;
  const int t    = threadIdx.x;
  const int lane = t & 31;
  const int wv   = t >> 5;
  __shared__ float colk[N_];
  __shared__ float l11[16 * 16];
  __shared__ float wredv[32];
  __shared__ int   wredi[32];
  __shared__ int   s_pivrow;
  float logsum = 0.f;
  int negpar = 0, zero = 0;

  for (int kb = 0; kb < n; kb += 16) {
    // ---------- panel factorization (columns kb..kb+15) ----------
    for (int k = kb; k < kb + 16; ++k) {
      // pivot search over column k, rows k..n-1 (wave shuffle + LDS tree)
      int i = k + t;
      float v  = (i < n) ? fabsf(Ab[(size_t)i * n + k]) : -1.f;
      int   vi = (i < n) ? i : n;
#pragma unroll
      for (int off = 16; off > 0; off >>= 1) {
        float ov = __shfl_xor(v, off, 32);
        int   oi = __shfl_xor(vi, off, 32);
        if (ov > v || (ov == v && oi < vi)) { v = ov; vi = oi; }
      }
      if (lane == 0) { wredv[wv] = v; wredi[wv] = vi; }
      __syncthreads();
      if (wv == 0) {
        float v2 = wredv[lane];
        int   i2 = wredi[lane];
#pragma unroll
        for (int off = 16; off > 0; off >>= 1) {
          float ov = __shfl_xor(v2, off, 32);
          int   oi = __shfl_xor(i2, off, 32);
          if (ov > v2 || (ov == v2 && oi < i2)) { v2 = ov; i2 = oi; }
        }
        if (lane == 0) s_pivrow = i2;
      }
      __syncthreads();
      int p = s_pivrow;
      if (p != k) {
        for (int j = t; j < n; j += 1024) {
          float tv = Ab[(size_t)k * n + j];
          Ab[(size_t)k * n + j] = Ab[(size_t)p * n + j];
          Ab[(size_t)p * n + j] = tv;
        }
        negpar ^= 1;
      }
      __syncthreads();
      float piv = Ab[(size_t)k * n + k];
      if (piv == 0.f) zero = 1;
      if (piv < 0.f) negpar ^= 1;
      logsum += logf(fabsf(piv));          // uniform across threads; deterministic
      float inv = (piv != 0.f) ? (1.f / piv) : 0.f;
      int i2 = k + 1 + t;
      if (i2 < n) {
        float l = Ab[(size_t)i2 * n + k] * inv;
        Ab[(size_t)i2 * n + k] = l;
        colk[i2] = l;
      }
      __syncthreads();
      // in-panel rank-1 update: columns k+1 .. kb+15 only
      if (k + 1 < kb + 16) {
        int i3 = k + 1 + t;
        if (i3 < n) {
          float li = colk[i3];
          for (int j = k + 1; j < kb + 16; ++j)
            Ab[(size_t)i3 * n + j] -= li * Ab[(size_t)k * n + j];
        }
      }
      __syncthreads();
    }
    if (kb + 16 >= n) break;   // last panel: no trailing matrix

    // ---------- stage diag block (multipliers) into LDS ----------
    if (t < 256) { int r = t >> 4, c = t & 15; l11[t] = Ab[(size_t)(kb + r) * n + kb + c]; }
    __syncthreads();

    // ---------- triangular solve: U12 = L11^{-1} A12  (unit lower) ----------
    for (int j = kb + 16 + t; j < n; j += 1024) {
      float u[16];
#pragma unroll
      for (int r = 0; r < 16; ++r) u[r] = Ab[(size_t)(kb + r) * n + j];
#pragma unroll
      for (int r = 1; r < 16; ++r)
        for (int s = 0; s < r; ++s) u[r] -= l11[r * 16 + s] * u[s];
#pragma unroll
      for (int r = 0; r < 16; ++r) Ab[(size_t)(kb + r) * n + j] = u[r];
    }
    __syncthreads();

    // ---------- trailing update A22 -= L21 * U12 via V_WMMA_F32_16X16X4_F32 ----------
    {
      const int m0  = kb + 16;
      const int tpr = (n - m0) >> 4;          // always a multiple of 16 remains
      const int ntiles = tpr * tpr;
      const int m = lane & 15, half = lane >> 4;
      for (int tix = wv; tix < ntiles; tix += 32) {   // wave-uniform => EXEC all 1s at WMMA
        int ti = tix / tpr, tj = tix - ti * tpr;
        int i0 = m0 + (ti << 4), j0 = m0 + (tj << 4);
        v8f c = {};
#pragma unroll
        for (int ks = 0; ks < 16; ks += 4) {
          int kc = kb + ks + half * 2;
          v2f a, bf;
          const float* arow = Ab + (size_t)(i0 + m) * n + kc;
          a[0]  = arow[0];
          a[1]  = arow[1];
          bf[0] = Ab[(size_t)kc * n + j0 + m];
          bf[1] = Ab[(size_t)(kc + 1) * n + j0 + m];
          c = __builtin_amdgcn_wmma_f32_16x16x4_f32(false, a, false, bf, (short)0, c, false, false);
        }
#pragma unroll
        for (int r = 0; r < 8; ++r) {
          size_t e = (size_t)(i0 + (half << 3) + r) * n + j0 + m;
          Ab[e] -= c[r];
        }
      }
    }
    __syncthreads();
  }
  if (t == 0) {
    logdetv[b] = logsum;
    signv[b]   = zero ? 0.f : (negpar ? -1.f : 1.f);
  }
}

// reduce gold partials deterministically + matrix-tree logistic loss
__global__ void finalize_kernel(const float* __restrict__ partials,
                                const float* __restrict__ rootsc,
                                const int* __restrict__ roots,
                                const float* __restrict__ logdetv,
                                const float* __restrict__ signv,
                                float* __restrict__ out) {
  __shared__ float gadj[B_];
  int t = threadIdx.x;
  if (t < B_) {
    const float* p = partials + (size_t)t * 1024;
    float s = 0.f;
    for (int i = 0; i < 1024; ++i) s += p[i];
    gadj[t] = s;
  }
  __syncthreads();
  if (t == 0) {
    const float kNaN = __builtin_nanf("");
    float total = 0.f;
    for (int b = 0; b < B_; ++b) {
      float gold = rootsc[(size_t)b * N_ + roots[b]] + gadj[b];
      float logdet = (signv[b] > 0.f) ? logdetv[b] : kNaN;
      bool gnan = isnan(gold), lnan = isnan(logdet);
      float valid = (!gnan && !lnan) ? 1.f : 0.f;
      float mask = (gold <= logdet * valid) ? 1.f : 0.f;
      float loss = (logdet - gold) * mask;
      if (isnan(loss)) loss = 0.f;
      total += loss;
    }
    out[0] = 0.25f * total / (float)B_;
  }
}

// ---------------- Launch ----------------
extern "C" void kernel_launch(void* const* d_in, const int* in_sizes, int n_in,
                              void* d_out, int out_size, void* d_ws, size_t ws_size,
                              hipStream_t stream) {
  (void)in_sizes; (void)n_in; (void)out_size; (void)ws_size;
  const float* h_cat   = (const float*)d_in[0];
  const float* left    = (const float*)d_in[1];
  const float* right   = (const float*)d_in[2];
  const int*   roots   = (const int*)  d_in[3];
  const float* W_bilin = (const float*)d_in[4];
  const float* b_bilin = (const float*)d_in[5];
  const float* W_head  = (const float*)d_in[6];
  const float* W_dep   = (const float*)d_in[7];
  const float* W_r1    = (const float*)d_in[8];
  const float* b_r1    = (const float*)d_in[9];
  const float* W_r2    = (const float*)d_in[10];
  const float* b_r2    = (const float*)d_in[11];
  float* out = (float*)d_out;

  char* ws = (char*)d_ws;
  size_t off = 0;
  auto carve = [&](size_t bytes) -> void* {
    void* p = ws + off;
    off += (bytes + 255) & ~(size_t)255;
    return p;
  };
  float*    lap      = (float*)   carve((size_t)B_ * N_ * N_ * 4);  // A, then Laplacian in place
  _Float16* h16      = (_Float16*)carve((size_t)B_ * N_ * H_ * 2);
  _Float16* tmp16    = (_Float16*)carve((size_t)B_ * 2 * N_ * H_ * 2);
  _Float16* wb16T    = (_Float16*)carve((size_t)2 * H_ * H_ * 2);
  _Float16* wr1_16   = (_Float16*)carve((size_t)H_ * H_ * 2);
  float*    head     = (float*)   carve((size_t)B_ * N_ * 2 * 4);
  float*    dep      = (float*)   carve((size_t)B_ * N_ * 2 * 4);
  float*    rootsc   = (float*)   carve((size_t)B_ * N_ * 4);
  float*    partials = (float*)   carve((size_t)B_ * 1024 * 4);
  float*    colsum   = (float*)   carve((size_t)B_ * N_ * 4);
  float*    logdetv  = (float*)   carve((size_t)B_ * 4);
  float*    signv    = (float*)   carve((size_t)B_ * 4);

  {
    int c1 = B_ * N_ * H_;
    f32_to_f16_kernel<<<(c1 + 255) / 256, 256, 0, stream>>>(h_cat, h16, c1);
    int c2 = 2 * H_ * H_;
    wbilin_transpose_f16_kernel<<<(c2 + 255) / 256, 256, 0, stream>>>(W_bilin, wb16T);
    int c3 = H_ * H_;
    f32_to_f16_kernel<<<(c3 + 255) / 256, 256, 0, stream>>>(W_r1, wr1_16, c3);
  }
  tmp_gemm_kernel<<<2048, 256, 0, stream>>>(h16, wb16T, tmp16);
  head_dep_kernel<<<(B_ * N_ + 255) / 256, 256, 0, stream>>>(h_cat, W_head, W_dep, head, dep);
  root_kernel<<<64, 256, 0, stream>>>(h16, wr1_16, b_r1, W_r2, b_r2, rootsc);
  compat_kernel<<<1024, 256, 0, stream>>>(tmp16, h16, head, dep, b_bilin, left, right, lap, partials);
  colsum_kernel<<<(B_ * N_ + 255) / 256, 256, 0, stream>>>(lap, colsum);
  build_lap_kernel<<<(B_ * N_ * N_) / 256, 256, 0, stream>>>(lap, colsum, rootsc);
  lu_kernel<<<B_, 1024, 0, stream>>>(lap, logdetv, signv);
  finalize_kernel<<<1, 256, 0, stream>>>(partials, rootsc, roots, logdetv, signv, out);
}
